// DeepwalkModel_17781164606023
// MI455X (gfx1250) — matrix operand order repaired and checked by reading, not compiled, not run on hardware
//
#include <hip/hip_runtime.h>

// ---------------------------------------------------------------------------
// Deepwalk hierarchical-softmax loss for MI455X (gfx1250, wave32).
//
// Gather-bound kernel (≈700MB of hsoftmax/emb row gathers vs 0.34 useful
// GFLOP -> ~30us roofline at 23.3 TB/s). One wave handles 16 batch elements.
// Per tree depth, the 16 gathered hsoftmax rows form the WMMA A matrix, the
// 16 v_emb rows (loaded once, reused 20x) form B, and the diagonal of
// D = A*B gives the 16 per-batch dot products. 4x v_wmma_f32_16x16x32_f16
// accumulate K=128 in f32.
// ---------------------------------------------------------------------------

typedef __attribute__((ext_vector_type(16))) _Float16 v16h;
typedef __attribute__((ext_vector_type(8)))  float    v8f;
typedef __attribute__((ext_vector_type(4)))  float    v4f;

#define NUM_LEAVES 1048576
#define EMB_DIM    128
#define TREE_DEPTH 20
#define TILE_B     16

// Pack 16 f32 -> v16h (f16); lowers to v_cvt_pk_f16_f32 pairs.
__device__ __forceinline__ v16h pack16(v4f a, v4f b, v4f c, v4f d) {
    v16h r;
    r[0]  = (_Float16)a[0]; r[1]  = (_Float16)a[1];
    r[2]  = (_Float16)a[2]; r[3]  = (_Float16)a[3];
    r[4]  = (_Float16)b[0]; r[5]  = (_Float16)b[1];
    r[6]  = (_Float16)b[2]; r[7]  = (_Float16)b[3];
    r[8]  = (_Float16)c[0]; r[9]  = (_Float16)c[1];
    r[10] = (_Float16)c[2]; r[11] = (_Float16)c[3];
    r[12] = (_Float16)d[0]; r[13] = (_Float16)d[1];
    r[14] = (_Float16)d[2]; r[15] = (_Float16)d[3];
    return r;
}

// A-matrix 16x32 f16 layout (ISA 7.12.2): lane l (row M = l&15):
//   lanes 0-15 : halves 0..7 -> K=kb+0..7,   halves 8..15 -> K=kb+16..23
//   lanes 16-31: halves 0..7 -> K=kb+8..15,  halves 8..15 -> K=kb+24..31
__device__ __forceinline__ v16h load_A_chunk(const float* __restrict__ row,
                                             int kb, int hi) {
    const v4f* p0 = (const v4f*)(row + kb + 8 * hi);
    const v4f* p1 = (const v4f*)(row + kb + 16 + 8 * hi);
    return pack16(p0[0], p0[1], p1[0], p1[1]);
}

// B-matrix 32x16 f16 layout (matches A): lane l (col N = l&15):
//   lanes 0-15 : halves 0..7 -> K=kb+0..7,   halves 8..15 -> K=kb+16..23
//   lanes 16-31: halves 0..7 -> K=kb+8..15,  halves 8..15 -> K=kb+24..31
__device__ __forceinline__ v16h load_B_chunk(const float* __restrict__ row,
                                             int kb, int hi) {
    const v4f* p0 = (const v4f*)(row + kb + 8 * hi);
    const v4f* p1 = (const v4f*)(row + kb + 16 + 8 * hi);
    return pack16(p0[0], p0[1], p1[0], p1[1]);
}

__global__ __launch_bounds__(256) void deepwalk_hsm_kernel(
    const float* __restrict__ emb_table,   // [V, 128]
    const float* __restrict__ hsoftmax,    // [V-1, 128]
    const int*   __restrict__ u,           // [B]
    const int*   __restrict__ v,           // [B]
    float*       __restrict__ out,         // [B]
    int n_tiles)
{
    const int lane = threadIdx.x & 31;
    const int wave = threadIdx.x >> 5;
    const int tile = blockIdx.x * (blockDim.x >> 5) + wave;
    if (tile >= n_tiles) return;           // wave-uniform: EXEC stays all-1s

    const int b0 = tile * TILE_B;
    const int r  = lane & 15;              // batch-in-tile this lane serves
    const int hi = lane >> 4;              // which half of the wave

    // --- B matrix: v_emb rows for the 16 batch elements, reused 20x ---
    const float* vrow = emb_table + (size_t)v[b0 + r] * EMB_DIM;
    v16h Bc0 = load_B_chunk(vrow, 0,  hi);
    v16h Bc1 = load_B_chunk(vrow, 32, hi);
    v16h Bc2 = load_B_chunk(vrow, 64, hi);
    v16h Bc3 = load_B_chunk(vrow, 96, hi);

    int node = (NUM_LEAVES - 1) + u[b0 + r];   // heap index of leaf

    // Diagonal ownership (C layout): (m,m) -> lane m, VGPR m (m<8)
    //                                      -> lane m+16, VGPR m-8 (m>=8)
    const bool valid = (lane < 8) || (lane >= 24);
    const int  cidx  = (lane < 8) ? lane : (lane - 24);

    float loss = 0.0f;

    #pragma unroll 4
    for (int d = 0; d < TREE_DEPTH; ++d) {
        const bool is_left = (node & 1) != 0;
        const int  parent  = (node - 1) >> 1;     // valid for both children
        const float* arow  = hsoftmax + (size_t)parent * EMB_DIM;

        // Prefetch next tree level (grandparent row) -> global_prefetch_b8
        if (d + 1 < TREE_DEPTH) {
            const int gp = (parent - 1) >> 1;
            __builtin_prefetch(hsoftmax + (size_t)gp * EMB_DIM + 64 * hi, 0, 1);
        }

        v8f C = {0.f, 0.f, 0.f, 0.f, 0.f, 0.f, 0.f, 0.f};
        {
            v16h A0 = load_A_chunk(arow, 0, hi);
            C = __builtin_amdgcn_wmma_f32_16x16x32_f16(false, A0, false, Bc0,
                                                       (short)0, C, false, false);
            v16h A1 = load_A_chunk(arow, 32, hi);
            C = __builtin_amdgcn_wmma_f32_16x16x32_f16(false, A1, false, Bc1,
                                                       (short)0, C, false, false);
            v16h A2 = load_A_chunk(arow, 64, hi);
            C = __builtin_amdgcn_wmma_f32_16x16x32_f16(false, A2, false, Bc2,
                                                       (short)0, C, false, false);
            v16h A3 = load_A_chunk(arow, 96, hi);
            C = __builtin_amdgcn_wmma_f32_16x16x32_f16(false, A3, false, Bc3,
                                                       (short)0, C, false, false);
        }

        // Branchless diagonal extraction (keeps EXEC full)
        float diag = C[0];
        diag = (cidx == 1) ? C[1] : diag;
        diag = (cidx == 2) ? C[2] : diag;
        diag = (cidx == 3) ? C[3] : diag;
        diag = (cidx == 4) ? C[4] : diag;
        diag = (cidx == 5) ? C[5] : diag;
        diag = (cidx == 6) ? C[6] : diag;
        diag = (cidx == 7) ? C[7] : diag;

        const float t = (is_left ? 1.0f : -1.0f) * diag;
        // -log_sigmoid(t) = max(-t,0) + log(1+exp(-|t|))  (stable; __logf/__expf
        // keep this to two trans ops instead of log1pf's rcp/division chain)
        const float nls = fmaxf(-t, 0.0f) + __logf(1.0f + __expf(-fabsf(t)));
        loss += valid ? nls : 0.0f;

        node = parent;
    }

    if (lane < 8)        out[b0 + lane]        = loss;
    else if (lane >= 24) out[b0 + (lane - 16)] = loss;
}

extern "C" void kernel_launch(void* const* d_in, const int* in_sizes, int n_in,
                              void* d_out, int out_size, void* d_ws, size_t ws_size,
                              hipStream_t stream) {
    const float* emb_table = (const float*)d_in[0];
    const float* hsoftmax  = (const float*)d_in[1];
    const int*   u         = (const int*)d_in[2];
    const int*   v         = (const int*)d_in[3];
    float*       out       = (float*)d_out;

    const int B = in_sizes[2];                 // 65536
    const int n_tiles = B / TILE_B;            // 4096 waves, 16 batch each
    const int waves_per_block = 8;             // 256 threads
    const int blocks = (n_tiles + waves_per_block - 1) / waves_per_block;

    deepwalk_hsm_kernel<<<blocks, waves_per_block * 32, 0, stream>>>(
        emb_table, hsoftmax, u, v, out, n_tiles);
}
